// Sampler_39797166965453
// MI455X (gfx1250) — matrix operand order repaired and checked by reading, not compiled, not run on hardware
//
#include <hip/hip_runtime.h>
#include <hip/hip_bf16.h>
#include <math.h>

// Problem sizes (fixed by the reference)
#define BB 256      // batch rows
#define DD 1024     // hidden dim
#define VV 32000    // vocab
#define TT 64       // history tokens per row

#define KSTRIDE 40  // LDS k-stride (elements): 80 B = 5x16 B -> aligned b128
                    // fragment reads AND conflict-free banks across 16 lanes

typedef __attribute__((ext_vector_type(16))) __bf16 v16bf;
typedef __attribute__((ext_vector_type(8)))  float  v8f;

union Frag { uint4 u[2]; v16bf v; };

__device__ inline unsigned short f32_to_bf16(float f) {
    unsigned u = __float_as_uint(f);
    u += 0x7FFFu + ((u >> 16) & 1u);          // round-to-nearest-even
    return (unsigned short)(u >> 16);
}
__device__ inline unsigned pack2bf16(float a, float b) {
    return (unsigned)f32_to_bf16(a) | ((unsigned)f32_to_bf16(b) << 16);
}

// ---------------------------------------------------------------------------
// Kernel 1: logits = (hidden @ emb^T) / temp   via v_wmma_f32_16x16x32_bf16
// 256 threads (8 waves) per workgroup; tile = all 256 rows x 64 vocab cols,
// so the 131 MB embedding streams from HBM exactly once; hidden (1 MB) stays
// L2-resident. Software-pipelined double-buffered LDS staging: global b128
// loads for chunk i+1 are in flight while chunk i's WMMAs execute.
// ---------------------------------------------------------------------------
__global__ __launch_bounds__(256) void gemm_logits_kernel(
    const float* __restrict__ hidden, const float* __restrict__ emb,
    const float* __restrict__ temps, float* __restrict__ logits)
{
    __shared__ __align__(16) unsigned short ldsH[2 * BB * KSTRIDE]; // 2 x 20.0 KB
    __shared__ __align__(16) unsigned short ldsE[2 * 64 * KSTRIDE]; // 2 x  5.0 KB
    __shared__ float invT[BB];

    const int tid     = threadIdx.x;
    const int lane    = tid & 31;
    const int wave    = tid >> 5;
    const int halfsel = lane >> 4;     // wave32 lane-half select
    const int l16     = lane & 15;
    const int nBase   = blockIdx.x * 64;
    const int wRow    = wave * 32;

    if (tid < BB) invT[tid] = 1.0f / temps[tid];

    v8f acc[2][4];
    for (int mt = 0; mt < 2; ++mt)
        for (int nt = 0; nt < 4; ++nt)
            for (int q = 0; q < 8; ++q) acc[mt][nt][q] = 0.0f;

    float4 hreg[8];   // 256 rows x 32 k  = 2048 float4 / 256 thr
    float4 ereg[2];   //  64 cols x 32 k  =  512 float4 / 256 thr

    // ---- global load of one k-chunk into registers (b128, coalescible) ----
    auto loadChunk = [&](int kc) {
#pragma unroll
        for (int i = 0; i < 8; ++i) {
            int idx4 = tid + i * 256;
            int row = idx4 >> 3, kk = (idx4 & 7) * 4;
            hreg[i] = *(const float4*)&hidden[row * DD + kc + kk];
        }
#pragma unroll
        for (int i = 0; i < 2; ++i) {
            int idx4 = tid + i * 256;
            int c = idx4 >> 3, kk = (idx4 & 7) * 4;
            ereg[i] = *(const float4*)&emb[(size_t)(nBase + c) * DD + kc + kk];
        }
    };
    // ---- convert f32 -> bf16 and stage into LDS buffer `buf` (b64 stores) ----
    auto stageChunk = [&](int buf) {
        unsigned short* H = ldsH + buf * (BB * KSTRIDE);
        unsigned short* E = ldsE + buf * (64 * KSTRIDE);
#pragma unroll
        for (int i = 0; i < 8; ++i) {
            int idx4 = tid + i * 256;
            int row = idx4 >> 3, kk = (idx4 & 7) * 4;
            uint2 p = make_uint2(pack2bf16(hreg[i].x, hreg[i].y),
                                 pack2bf16(hreg[i].z, hreg[i].w));
            *(uint2*)&H[row * KSTRIDE + kk] = p;
        }
#pragma unroll
        for (int i = 0; i < 2; ++i) {
            int idx4 = tid + i * 256;
            int c = idx4 >> 3, kk = (idx4 & 7) * 4;
            uint2 p = make_uint2(pack2bf16(ereg[i].x, ereg[i].y),
                                 pack2bf16(ereg[i].z, ereg[i].w));
            *(uint2*)&E[c * KSTRIDE + kk] = p;
        }
    };

    loadChunk(0);
    stageChunk(0);
    int cur = 0;

    for (int ck = 0; ck < DD / 32; ++ck) {
        int knext = (ck + 1) * 32;
        if (knext < DD) loadChunk(knext);          // in flight during WMMAs
        __syncthreads();                            // buf[cur] fully staged

        const unsigned short* H = ldsH + cur * (BB * KSTRIDE);
        const unsigned short* E = ldsE + cur * (64 * KSTRIDE);

        // B fragments (32x16 KxN): lane<16 -> N=l16,K 0..15; lane>=16 -> K 16..31
        Frag bF[4];
#pragma unroll
        for (int nt = 0; nt < 4; ++nt) {
            int base = (nt * 16 + l16) * KSTRIDE + halfsel * 16;
            bF[nt].u[0] = *(const uint4*)&E[base];
            bF[nt].u[1] = *(const uint4*)&E[base + 8];
        }
        // A fragments (16x32 MxK): lane<16 -> M=l16,K 0..15;
        // lane>=16 -> M=l16,K 16..31
#pragma unroll
        for (int mt = 0; mt < 2; ++mt) {
            Frag aF;
            int base = (wRow + mt * 16 + l16) * KSTRIDE;
            aF.u[0] = *(const uint4*)&H[base + halfsel * 16];
            aF.u[1] = *(const uint4*)&H[base + halfsel * 16 + 8];
#pragma unroll
            for (int nt = 0; nt < 4; ++nt) {
                acc[mt][nt] = __builtin_amdgcn_wmma_f32_16x16x32_bf16(
                    false, aF.v, false, bF[nt].v,
                    (short)0, acc[mt][nt], false, false);
            }
        }
        if (knext < DD) stageChunk(cur ^ 1);       // waits on global loads here
        cur ^= 1;
    }
    __syncthreads();

    // C/D layout: lanes 0-15: N=lane, M=vgpr; lanes 16-31: N=lane-16, M=vgpr+8
    for (int mt = 0; mt < 2; ++mt) {
        int mBase = wRow + mt * 16 + halfsel * 8;
        for (int nt = 0; nt < 4; ++nt) {
            int col = nBase + nt * 16 + l16;
            for (int j = 0; j < 8; ++j) {
                int m = mBase + j;
                logits[(size_t)m * VV + col] = acc[mt][nt][j] * invT[m];
            }
        }
    }
}

// ---------------------------------------------------------------------------
// Kernel 2: presence/frequency penalties as sparse atomic updates.
// logits already divided by temp, so deltas are scaled by 1/temp too.
// ---------------------------------------------------------------------------
__global__ __launch_bounds__(64) void penalty_kernel(
    const int* __restrict__ toks, const float* __restrict__ pres,
    const float* __restrict__ freq, const float* __restrict__ temps,
    float* __restrict__ logits)
{
    int b = blockIdx.x, t = threadIdx.x;
    int tok = toks[b * TT + t];
    bool first = true;
    for (int s = 0; s < t; ++s)
        if (toks[b * TT + s] == tok) { first = false; break; }
    float iT = 1.0f / temps[b];
    float delta = -(freq[b] + (first ? pres[b] : 0.0f)) * iT;
    atomicAdd(&logits[(size_t)b * VV + tok], delta);
}

// ---------------------------------------------------------------------------
// Kernel 3: softmax + top-k/top-p filtering via 3-level radix select on the
// float bit pattern of e = exp(logit - max). Kept set is {bits >= U} with
// U = max(U_topk, U_topp). One block (1024 thr = 32 waves) per row; the row
// lives in registers (32 values/thread, fully unrolled, float4 vmem).
// ---------------------------------------------------------------------------
#define NB1 4096
#define NB3 128

__device__ inline float waveMax(float v) {
    for (int o = 16; o; o >>= 1) v = fmaxf(v, __shfl_xor(v, o, 32));
    return v;
}
__device__ inline float waveSum(float v) {
    for (int o = 16; o; o >>= 1) v += __shfl_xor(v, o, 32);
    return v;
}
__device__ float blockMax(float v, float* sc) {
    int w = threadIdx.x >> 5, l = threadIdx.x & 31;
    v = waveMax(v);
    if (l == 0) sc[w] = v;
    __syncthreads();
    if (w == 0) { float x = waveMax(sc[l]); if (l == 0) sc[32] = x; }
    __syncthreads();
    float r = sc[32];
    __syncthreads();
    return r;
}
__device__ float blockSum(float v, float* sc) {
    int w = threadIdx.x >> 5, l = threadIdx.x & 31;
    v = waveSum(v);
    if (l == 0) sc[w] = v;
    __syncthreads();
    if (w == 0) { float x = waveSum(sc[l]); if (l == 0) sc[32] = x; }
    __syncthreads();
    float r = sc[32];
    __syncthreads();
    return r;
}

// Find bin b (descending) where count of strictly higher bins < kTgt <= +cnt[b].
__device__ void crossCount(const unsigned* cnt, int nbins, unsigned kTgt,
                           unsigned* sc, unsigned* resBin, unsigned* resRem)
{
    int tid = threadIdx.x, chunk = nbins >> 5;
    if (tid < 32) {
        unsigned c = 0;
        for (int b = tid * chunk, e2 = b + chunk; b < e2; ++b) c += cnt[b];
        sc[tid] = c;
    }
    __syncthreads();
    if (tid == 0) {
        unsigned above = 0, bin = 0, rem = 1; int found = 0;
        for (int j = 31; j >= 0 && !found; --j) {
            if (above + sc[j] >= kTgt) {
                for (int b = (j + 1) * chunk - 1; b >= j * chunk; --b) {
                    if (above + cnt[b] >= kTgt) { bin = (unsigned)b; rem = kTgt - above; found = 1; break; }
                    above += cnt[b];
                }
            } else above += sc[j];
        }
        *resBin = found ? bin : 0u;
        *resRem = found ? rem : 1u;
    }
    __syncthreads();
}

// Find bin b (descending) where sum of strictly higher bins <= pTgt < +sum[b].
__device__ void crossSum(const float* sums, int nbins, float pTgt,
                         float* sc, unsigned* resBin, float* resRem)
{
    int tid = threadIdx.x, chunk = nbins >> 5;
    if (tid < 32) {
        float s = 0.f;
        for (int b = tid * chunk, e2 = b + chunk; b < e2; ++b) s += sums[b];
        sc[tid] = s;
    }
    __syncthreads();
    if (tid == 0) {
        float above = 0.f, rem = 3.0e38f; unsigned bin = 0; int found = 0;
        for (int j = 31; j >= 0 && !found; --j) {
            if (above + sc[j] > pTgt) {
                for (int b = (j + 1) * chunk - 1; b >= j * chunk; --b) {
                    if (above + sums[b] > pTgt) { bin = (unsigned)b; rem = pTgt - above; found = 1; break; }
                    above += sums[b];
                }
            } else above += sc[j];
        }
        *resBin = found ? bin : 0u;      // not found => top_p covers all: keep everything
        *resRem = found ? rem : 3.0e38f;
    }
    __syncthreads();
}

__global__ __launch_bounds__(1024) void sampler_kernel(
    float* __restrict__ probs,                 // in: penalized logits/T; out: filtered probs
    const float* __restrict__ top_ps, const int* __restrict__ top_ks)
{
    __shared__ unsigned cntA[NB1];
    __shared__ float    sumA[NB1];
    __shared__ float    sumB[NB1];
    __shared__ float    scF[33];
    __shared__ unsigned scU[33];
    __shared__ unsigned sBin; __shared__ unsigned sRemU; __shared__ float sRemF;

    const int r = blockIdx.x, tid = threadIdx.x;
    float* row = probs + (size_t)r * VV;

    float evr[32];
    float lmax = -3.0e38f;
#pragma unroll
    for (int j = 0; j < 8; ++j) {                 // float4-vectorized row load
        int i4 = tid + j * 1024;
        if (i4 < VV / 4) {
            float4 x = *(const float4*)&row[i4 * 4];
            evr[4*j+0] = x.x; evr[4*j+1] = x.y; evr[4*j+2] = x.z; evr[4*j+3] = x.w;
            lmax = fmaxf(fmaxf(fmaxf(lmax, x.x), fmaxf(x.y, x.z)), x.w);
        }
    }
    float M = blockMax(lmax, scF);

    for (int i = tid; i < NB1; i += 1024) { cntA[i] = 0u; sumA[i] = 0.f; }
    __syncthreads();

    float lsum = 0.f;
#pragma unroll
    for (int j = 0; j < 8; ++j) {
        int i4 = tid + j * 1024;
        if (i4 < VV / 4) {
#pragma unroll
            for (int c = 0; c < 4; ++c) {
                float ex = __expf(evr[4*j+c] - M);
                evr[4*j+c] = ex; lsum += ex;
                unsigned u = __float_as_uint(ex);  // positive float: bit order = value order
                atomicAdd(&cntA[u >> 19], 1u);
                atomicAdd(&sumA[u >> 19], ex);
            }
        }
    }
    float S = blockSum(lsum, scF);
    unsigned k = (unsigned)top_ks[r];
    float    P = top_ps[r] * S;

    // ---- level 1: bits[30:19] ----
    crossCount(cntA, NB1, k, scU, &sBin, &sRemU);
    unsigned b1K = sBin, remK1 = sRemU;
    crossSum(sumA, NB1, P, scF, &sBin, &sRemF);
    unsigned b1P = sBin; float remP1 = sRemF;

    // ---- level 2: bits[18:7] within each chain's bin ----
    for (int i = tid; i < NB1; i += 1024) { cntA[i] = 0u; sumB[i] = 0.f; }
    __syncthreads();
#pragma unroll
    for (int j = 0; j < 8; ++j) {
        int i4 = tid + j * 1024;
        if (i4 < VV / 4) {
#pragma unroll
            for (int c = 0; c < 4; ++c) {
                unsigned u = __float_as_uint(evr[4*j+c]);
                unsigned hi = u >> 19;
                if (hi == b1K) atomicAdd(&cntA[(u >> 7) & 0xFFFu], 1u);
                if (hi == b1P) atomicAdd(&sumB[(u >> 7) & 0xFFFu], evr[4*j+c]);
            }
        }
    }
    __syncthreads();
    crossCount(cntA, NB1, remK1, scU, &sBin, &sRemU);
    unsigned b2K = sBin, remK2 = sRemU;
    crossSum(sumB, NB1, remP1, scF, &sBin, &sRemF);
    unsigned b2P = sBin; float remP2 = sRemF;
    unsigned preK = (b1K << 12) | b2K;
    unsigned preP = (b1P << 12) | b2P;

    // ---- level 3: bits[6:0] ----
    if (tid < NB3) { cntA[tid] = 0u; sumB[tid] = 0.f; }
    __syncthreads();
#pragma unroll
    for (int j = 0; j < 8; ++j) {
        int i4 = tid + j * 1024;
        if (i4 < VV / 4) {
#pragma unroll
            for (int c = 0; c < 4; ++c) {
                unsigned u = __float_as_uint(evr[4*j+c]);
                if ((u >> 7) == preK) atomicAdd(&cntA[u & 127u], 1u);
                if ((u >> 7) == preP) atomicAdd(&sumB[u & 127u], evr[4*j+c]);
            }
        }
    }
    __syncthreads();
    crossCount(cntA, NB3, remK2, scU, &sBin, &sRemU);
    unsigned b3K = sBin;
    crossSum(sumB, NB3, remP2, scF, &sBin, &sRemF);
    unsigned b3P = sBin;

    unsigned Uk = (preK << 7) | b3K;
    unsigned Up = (preP << 7) | b3P;
    unsigned U  = (Uk > Up) ? Uk : Up;

    float invS = 1.0f / S;
#pragma unroll
    for (int j = 0; j < 8; ++j) {                 // float4-vectorized store
        int i4 = tid + j * 1024;
        if (i4 < VV / 4) {
            float4 o;
            o.x = (__float_as_uint(evr[4*j+0]) >= U) ? evr[4*j+0] * invS : 0.0f;
            o.y = (__float_as_uint(evr[4*j+1]) >= U) ? evr[4*j+1] * invS : 0.0f;
            o.z = (__float_as_uint(evr[4*j+2]) >= U) ? evr[4*j+2] * invS : 0.0f;
            o.w = (__float_as_uint(evr[4*j+3]) >= U) ? evr[4*j+3] * invS : 0.0f;
            *(float4*)&row[i4 * 4] = o;
        }
    }
}

// ---------------------------------------------------------------------------
extern "C" void kernel_launch(void* const* d_in, const int* in_sizes, int n_in,
                              void* d_out, int out_size, void* d_ws, size_t ws_size,
                              hipStream_t stream)
{
    (void)in_sizes; (void)n_in; (void)out_size; (void)d_ws; (void)ws_size;
    const float* hidden = (const float*)d_in[0];
    const float* emb    = (const float*)d_in[1];
    const int*   toks   = (const int*)d_in[2];
    const float* pres   = (const float*)d_in[3];
    const float* freq   = (const float*)d_in[4];
    const float* temps  = (const float*)d_in[5];
    const float* topp   = (const float*)d_in[6];
    const int*   topk   = (const int*)d_in[7];
    float* out = (float*)d_out;   // used as logits scratch, then final probs

    gemm_logits_kernel<<<VV / 64, 256, 0, stream>>>(hidden, emb, temps, out);
    penalty_kernel<<<BB, TT, 0, stream>>>(toks, pres, freq, temps, out);
    sampler_kernel<<<BB, 1024, 0, stream>>>(out, topp, topk);
}
